// NNConvModel_52621939310755
// MI455X (gfx1250) — matrix operand order repaired and checked by reading, not compiled, not run on hardware
//
#include <hip/hip_runtime.h>
#include <hip/hip_bf16.h>

typedef __attribute__((ext_vector_type(16))) _Float16 v16h;
typedef __attribute__((ext_vector_type(8)))  float    v8f;
typedef __attribute__((ext_vector_type(4)))  unsigned int v4u;
typedef __attribute__((ext_vector_type(4)))  int v4i;

union V16H { v16h h; v4u q[2]; };

#if defined(__has_builtin)
#if __has_builtin(__builtin_amdgcn_global_load_async_to_lds_b128) && \
    __has_builtin(__builtin_amdgcn_s_wait_asynccnt)
#define USE_ASYNC_LDS 1
#endif
#endif

#ifdef USE_ASYNC_LDS
typedef __attribute__((address_space(1))) v4i* g_v4i_ptr;
typedef __attribute__((address_space(3))) v4i* l_v4i_ptr;
#endif

__device__ __forceinline__ float lrelu_f(float v) { return v > 0.f ? v : 0.1f * v; }

// Stage nhalf f16 elements (multiple of 8) from global to LDS, all threads of
// a 256-thread block cooperating.  Uses CDNA5 async global->LDS loads
// (ASYNCcnt) when available.  Caller must __syncthreads() afterwards.
__device__ __forceinline__ void stage_to_lds(const _Float16* __restrict__ g,
                                             _Float16* l, int nhalf)
{
    for (int c = threadIdx.x * 8; c < nhalf; c += 256 * 8) {
#ifdef USE_ASYNC_LDS
        __builtin_amdgcn_global_load_async_to_lds_b128(
            (g_v4i_ptr)(g + c), (l_v4i_ptr)(l + c), 0, 0);
#else
        *(v4u*)(l + c) = *(const v4u*)(g + c);
#endif
    }
#ifdef USE_ASYNC_LDS
    __builtin_amdgcn_s_wait_asynccnt(0);
#endif
}

// ---------------------------------------------------------------------------
// BatchNorm statistics: one block computes per-column affine (a, b) such that
// y = x*a + b  ==  (x - mean) * rsqrt(var+eps) * g + beta
// ---------------------------------------------------------------------------
__global__ __launch_bounds__(256) void bn_stats_kernel(
    const float* __restrict__ x, int rows, int C,
    const float* __restrict__ g, const float* __restrict__ beta,
    float* __restrict__ oa, float* __restrict__ ob)
{
    __shared__ float ss[32];
    __shared__ float sq[32];
    int tid = threadIdx.x;
    if (tid < 32) { ss[tid] = 0.f; sq[tid] = 0.f; }
    __syncthreads();
    int stride = 256 / C;            // threads per column
    int c  = tid % C;
    int r0 = tid / C;
    if (r0 < stride) {
        float s = 0.f, q = 0.f;
        for (int r = r0; r < rows; r += stride) {
            float v = x[(size_t)r * C + c];
            s += v; q += v * v;
        }
        atomicAdd(&ss[c], s);
        atomicAdd(&sq[c], q);
    }
    __syncthreads();
    if (tid < C) {
        float m   = ss[tid] / (float)rows;
        float var = sq[tid] / (float)rows - m * m;   // biased variance
        float a   = g[tid] * rsqrtf(var + 1e-5f);
        oa[tid] = a;
        ob[tid] = beta[tid] - m * a;
    }
}

__global__ __launch_bounds__(256) void affine_kernel(
    const float* __restrict__ x, float* __restrict__ y, int n, int C,
    const float* __restrict__ a, const float* __restrict__ b)
{
    int i = blockIdx.x * 256 + threadIdx.x;
    if (i >= n) return;
    int c = i % C;
    y[i] = x[i] * a[c] + b[c];
}

// ---------------------------------------------------------------------------
// Pack f32 weight [Ksrc x Ncols] into f16 WMMA B-fragment blocks.
// Layout: [(kc*NT + nt)*16 + n]*32 + k  ->  each lane of a wave reads one
// contiguous 32B span of a 1KB tile (bank-conflict-free from LDS).
// ---------------------------------------------------------------------------
__global__ __launch_bounds__(256) void pack_b_kernel(
    const float* __restrict__ W, _Float16* __restrict__ out,
    int Ksrc, int Ncols, int NT, int total)
{
    int idx = blockIdx.x * 256 + threadIdx.x;
    if (idx >= total) return;
    int k  = idx & 31;
    int n  = (idx >> 5) & 15;
    int nt = (idx >> 9) % NT;
    int kc = idx / (512 * NT);
    int K   = kc * 32 + k;
    int col = nt * 16 + n;
    float v = (K < Ksrc) ? W[(size_t)K * Ncols + col] : 0.f;
    out[idx] = (_Float16)v;
}

// ---------------------------------------------------------------------------
// agg[n][o] = bias[o] + sum_i x[n][i] * root[i][o]   (initializes scatter dst)
// ---------------------------------------------------------------------------
template<int CIN, int COUT>
__global__ __launch_bounds__(256) void root_kernel(
    const float* __restrict__ x, const float* __restrict__ root,
    const float* __restrict__ bias, float* __restrict__ agg, int N)
{
    int idx = blockIdx.x * 256 + threadIdx.x;
    if (idx >= N * COUT) return;
    int n = idx / COUT, o = idx % COUT;
    float s = bias[o];
    #pragma unroll
    for (int i = 0; i < CIN; ++i) s += x[(size_t)n * CIN + i] * root[i * COUT + o];
    agg[idx] = s;
}

// ---------------------------------------------------------------------------
// Fused NNConv edge kernel.  Per 16-edge wave tile:
//   h  = lrelu(en @ W1 + b1)                (VALU, tiny: 10->CIN)
//   A  = f16(h)  [16 x 32]  (K zero-padded for CIN=16)
//   for each 16-col tile t of W2:   D = A x Bs[t]   (v_wmma_f32_16x16x32_f16)
//       w = lrelu(D + b2);  msg[m][o] += x[src[m]][i] * w    (i = 16t/COUT)
//   atomicAdd msg into agg[dst] (L2-resident table)
// Packed W2 is staged once per block into LDS (async global->LDS), so the
// dominant B-fragment stream is served at LDS bandwidth instead of 1GB of
// repeated L2 reads.  The E x (CIN*COUT) weight tensor (1GB f32 for layer 2)
// is never materialized.
// ---------------------------------------------------------------------------
template<int CIN, int COUT>
__global__ __launch_bounds__(256) void nnconv_kernel(
    const float* __restrict__ en,       // [E,10] normalized edge attrs
    const float* __restrict__ xfeat,    // [N,CIN] node features
    const int*   __restrict__ ei,       // [2,E]
    const float* __restrict__ W1,       // [10,CIN]
    const float* __restrict__ b1,       // [CIN]
    const _Float16* __restrict__ Bp,    // packed f16 W2 fragments (global)
    const float* __restrict__ b2,       // [CIN*COUT]
    float* __restrict__ agg,            // [N,COUT] pre-initialized with root
    int E)
{
    constexpr int NT      = COUT / 16;
    constexpr int NHALF_B = CIN * COUT * 32;   // f16 elements of packed W2
    __shared__ _Float16 Bs[NHALF_B];           // 32KB (L1) / 128KB (L2)
    __shared__ _Float16 hA[8][16 * 32];
    __shared__ float    xs[8][16 * CIN];
    __shared__ int      dstid[8][16];

    stage_to_lds(Bp, Bs, NHALF_B);

    const int w    = threadIdx.x >> 5;
    const int l    = threadIdx.x & 31;
    const int m    = l & 15;
    const int hgrp = l >> 4;
    const int tile = blockIdx.x * 8 + w;
    const int e    = tile * 16 + m;

    if (hgrp == 0) {
        // lanes 0-15: edge-weight MLP hidden layer for edge m
        float ev[10];
        #pragma unroll
        for (int j = 0; j < 10; ++j) ev[j] = en[(size_t)e * 10 + j];
        for (int i = 0; i < CIN; ++i) {
            float s = b1[i];
            #pragma unroll
            for (int j = 0; j < 10; ++j) s += ev[j] * W1[j * CIN + i];
            hA[w][m * 32 + i] = (_Float16)lrelu_f(s);
        }
        for (int i = CIN; i < 32; ++i) hA[w][m * 32 + i] = (_Float16)0.f;  // K pad
    } else {
        // lanes 16-31: gather source node features + destination ids
        int s = ei[e];
        for (int c = 0; c < CIN; ++c) xs[w][m * CIN + c] = xfeat[(size_t)s * CIN + c];
        dstid[w][m] = ei[E + e];
    }
    __syncthreads();

    // A fragment: lane holds row M=l%16; K=16*hgrp..+15 contiguous
    V16H a;
    a.q[0] = *(const v4u*)&hA[w][m * 32 + 16 * hgrp];
    a.q[1] = *(const v4u*)&hA[w][m * 32 + 16 * hgrp + 8];

    float macc[NT][8] = {};

    for (int i = 0; i < CIN; ++i) {
        float xv[8];
        #pragma unroll
        for (int r = 0; r < 8; ++r) xv[r] = xs[w][(r + 8 * hgrp) * CIN + i];
        #pragma unroll
        for (int ot = 0; ot < NT; ++ot) {
            const int t = i * NT + ot;   // flat 16-column tile of W2
            V16H b;
            const v4u* gp = (const v4u*)&Bs[(size_t)t * 512 + m * 32 + hgrp * 16];
            b.q[0] = gp[0];
            b.q[1] = gp[1];
            v8f c = {};
            c = __builtin_amdgcn_wmma_f32_16x16x32_f16(
                    false, a.h, false, b.h, (short)0, c, false, false);
            const float bb = b2[t * 16 + m];
            #pragma unroll
            for (int r = 0; r < 8; ++r) {
                float v = lrelu_f(c[r] + bb);      // per-edge weight element
                macc[ot][r] += xv[r] * v;          // message accumulation
            }
        }
    }

    // scatter-add messages: element (edge M=r+8*hgrp, out col ot*16+m)
    #pragma unroll
    for (int ot = 0; ot < NT; ++ot) {
        const int o = ot * 16 + m;
        #pragma unroll
        for (int r = 0; r < 8; ++r) {
            const int mm = r + 8 * hgrp;
            atomicAdd(&agg[(size_t)dstid[w][mm] * COUT + o], macc[ot][r]);
        }
    }
}

// ---------------------------------------------------------------------------
// Edge predictor: h = [x2[src] || x2[dst] || en] (138, padded to 160)
// layer1 (138->64) via 5-chunk WMMA with LDS-staged B; tail MLP in VALU.
// ---------------------------------------------------------------------------
__global__ __launch_bounds__(256) void edge_mlp_kernel(
    const float* __restrict__ x2, const float* __restrict__ en,
    const int* __restrict__ ei, const _Float16* __restrict__ B1p,
    const float* __restrict__ eb1,
    const float* __restrict__ ew2, const float* __restrict__ eb2,
    const float* __restrict__ ew3, const float* __restrict__ eb3,
    const float* __restrict__ ew4, const float* __restrict__ eb4,
    const float* __restrict__ ew5, const float* __restrict__ eb5,
    float* __restrict__ out, int E)
{
    __shared__ _Float16 Bs[20 * 512];          // 20KB: packed ew1 fragments
    __shared__ _Float16 fA[8][16 * 160];
    __shared__ float    h1[8][16 * 64];

    stage_to_lds(B1p, Bs, 20 * 512);

    const int w    = threadIdx.x >> 5;
    const int l    = threadIdx.x & 31;
    const int m    = l & 15;
    const int hgrp = l >> 4;
    const int tile = blockIdx.x * 8 + w;
    const int e    = tile * 16 + m;
    const int s    = ei[e];
    const int d    = ei[E + e];

    if (hgrp == 0) {
        for (int k = 0; k < 64; ++k) fA[w][m * 160 + k] = (_Float16)x2[(size_t)s * 64 + k];
        #pragma unroll
        for (int j = 0; j < 10; ++j) fA[w][m * 160 + 128 + j] = (_Float16)en[(size_t)e * 10 + j];
        #pragma unroll
        for (int j = 0; j < 22; ++j) fA[w][m * 160 + 138 + j] = (_Float16)0.f;  // K pad
    } else {
        for (int k = 0; k < 64; ++k) fA[w][m * 160 + 64 + k] = (_Float16)x2[(size_t)d * 64 + k];
    }
    __syncthreads();

    V16H a[5];
    #pragma unroll
    for (int kc = 0; kc < 5; ++kc) {
        const int base = m * 160 + kc * 32;
        a[kc].q[0] = *(const v4u*)&fA[w][base + 16 * hgrp];
        a[kc].q[1] = *(const v4u*)&fA[w][base + 16 * hgrp + 8];
    }

    #pragma unroll
    for (int nt = 0; nt < 4; ++nt) {
        v8f c = {};
        #pragma unroll
        for (int kc = 0; kc < 5; ++kc) {
            V16H b;
            const v4u* gp = (const v4u*)&Bs[(size_t)((kc * 4 + nt) * 16 + m) * 32 + hgrp * 16];
            b.q[0] = gp[0];
            b.q[1] = gp[1];
            c = __builtin_amdgcn_wmma_f32_16x16x32_f16(
                    false, a[kc].h, false, b.h, (short)0, c, false, false);
        }
        const float bb = eb1[nt * 16 + m];
        #pragma unroll
        for (int r = 0; r < 8; ++r)
            h1[w][(r + 8 * hgrp) * 64 + nt * 16 + m] = lrelu_f(c[r] + bb);
    }
    __syncthreads();

    if (hgrp == 0) {
        float hr[64];
        const float4* hp = (const float4*)&h1[w][m * 64];
        #pragma unroll
        for (int k = 0; k < 16; ++k) {
            float4 v = hp[k];
            hr[4 * k] = v.x; hr[4 * k + 1] = v.y; hr[4 * k + 2] = v.z; hr[4 * k + 3] = v.w;
        }
        float h2v[32];
        for (int o = 0; o < 32; ++o) {
            float sacc = eb2[o];
            for (int k = 0; k < 64; ++k) sacc += hr[k] * ew2[k * 32 + o];
            h2v[o] = lrelu_f(sacc);
        }
        float h3v[16];
        for (int o = 0; o < 16; ++o) {
            float sacc = eb3[o];
            for (int k = 0; k < 32; ++k) sacc += h2v[k] * ew3[k * 16 + o];
            h3v[o] = lrelu_f(sacc);
        }
        float h4v[8];
        #pragma unroll
        for (int o = 0; o < 8; ++o) {
            float sacc = eb4[o];
            #pragma unroll
            for (int k = 0; k < 16; ++k) sacc += h3v[k] * ew4[k * 8 + o];
            h4v[o] = lrelu_f(sacc);
        }
        float o0 = eb5[0], o1 = eb5[1];
        #pragma unroll
        for (int k = 0; k < 8; ++k) { o0 += h4v[k] * ew5[k * 2]; o1 += h4v[k] * ew5[k * 2 + 1]; }
        out[(size_t)e * 2]     = o0;
        out[(size_t)e * 2 + 1] = o1;
    }
}

// ---------------------------------------------------------------------------
extern "C" void kernel_launch(void* const* d_in, const int* in_sizes, int n_in,
                              void* d_out, int out_size, void* d_ws, size_t ws_size,
                              hipStream_t stream)
{
    const float* x        = (const float*)d_in[0];
    const float* e        = (const float*)d_in[1];
    const int*   ei       = (const int*)d_in[2];
    const float* bn_ng    = (const float*)d_in[4];
    const float* bn_nb    = (const float*)d_in[5];
    const float* bn_eg    = (const float*)d_in[6];
    const float* bn_eb    = (const float*)d_in[7];
    const float* nn1_W1   = (const float*)d_in[8];
    const float* nn1_b1   = (const float*)d_in[9];
    const float* nn1_W2   = (const float*)d_in[10];
    const float* nn1_b2   = (const float*)d_in[11];
    const float* root1    = (const float*)d_in[12];
    const float* bias1    = (const float*)d_in[13];
    const float* nn2_W1   = (const float*)d_in[14];
    const float* nn2_b1   = (const float*)d_in[15];
    const float* nn2_W2   = (const float*)d_in[16];
    const float* nn2_b2   = (const float*)d_in[17];
    const float* root2    = (const float*)d_in[18];
    const float* bias2    = (const float*)d_in[19];
    const float* ew1      = (const float*)d_in[20];
    const float* eb1      = (const float*)d_in[21];
    const float* ew2      = (const float*)d_in[22];
    const float* eb2      = (const float*)d_in[23];
    const float* ew3      = (const float*)d_in[24];
    const float* eb3      = (const float*)d_in[25];
    const float* ew4      = (const float*)d_in[26];
    const float* eb4      = (const float*)d_in[27];
    const float* ew5      = (const float*)d_in[28];
    const float* eb5      = (const float*)d_in[29];
    float* out = (float*)d_out;

    const int N = in_sizes[0] / 16;   // 32768
    const int E = in_sizes[1] / 10;   // 131072

    // Workspace layout (~20.1 MB)
    char* base = (char*)d_ws;
    size_t off = 0;
    auto take = [&](size_t bytes) { void* p = base + off; off = (off + bytes + 255) & ~(size_t)255; return p; };
    float*    bnp = (float*)take(64 * sizeof(float));   // a_x[16],b_x[16],a_e[16],b_e[16]
    float*    xn  = (float*)take((size_t)N * 16 * 4);
    float*    enr = (float*)take((size_t)E * 10 * 4);
    float*    x1  = (float*)take((size_t)N * 32 * 4);
    float*    x2  = (float*)take((size_t)N * 64 * 4);
    _Float16* p1  = (_Float16*)take((size_t)1 * 32  * 512 * 2);  // nn1_W2 packed
    _Float16* p2  = (_Float16*)take((size_t)1 * 128 * 512 * 2);  // nn2_W2 packed
    _Float16* pE  = (_Float16*)take((size_t)5 * 4   * 512 * 2);  // ew1 packed (K->160)
    (void)ws_size;

    float* a_x = bnp + 0;  float* b_x = bnp + 16;
    float* a_e = bnp + 32; float* b_e = bnp + 48;

    // 1. BatchNorm affine params + apply
    bn_stats_kernel<<<1, 256, 0, stream>>>(x, N, 16, bn_ng, bn_nb, a_x, b_x);
    bn_stats_kernel<<<1, 256, 0, stream>>>(e, E, 10, bn_eg, bn_eb, a_e, b_e);
    affine_kernel<<<(N * 16 + 255) / 256, 256, 0, stream>>>(x, xn, N * 16, 16, a_x, b_x);
    affine_kernel<<<(E * 10 + 255) / 256, 256, 0, stream>>>(e, enr, E * 10, 10, a_e, b_e);

    // 2. Pack WMMA B fragments (f32 -> f16, fragment-order, K zero-padded)
    pack_b_kernel<<<(16384 + 255) / 256, 256, 0, stream>>>(nn1_W2, p1, 16, 512, 32, 16384);
    pack_b_kernel<<<(65536 + 255) / 256, 256, 0, stream>>>(nn2_W2, p2, 32, 2048, 128, 65536);
    pack_b_kernel<<<(10240 + 255) / 256, 256, 0, stream>>>(ew1, pE, 138, 64, 4, 10240);

    // 3. NNConv layer 1: x1 = root-init then edge scatter
    root_kernel<16, 32><<<(N * 32 + 255) / 256, 256, 0, stream>>>(xn, root1, bias1, x1, N);
    nnconv_kernel<16, 32><<<E / 128, 256, 0, stream>>>(enr, xn, ei, nn1_W1, nn1_b1, p1, nn1_b2, x1, E);

    // 4. NNConv layer 2
    root_kernel<32, 64><<<(N * 64 + 255) / 256, 256, 0, stream>>>(x1, root2, bias2, x2, N);
    nnconv_kernel<32, 64><<<E / 128, 256, 0, stream>>>(enr, x1, ei, nn2_W1, nn2_b1, p2, nn2_b2, x2, E);

    // 5. Edge predictor MLP
    edge_mlp_kernel<<<E / 128, 256, 0, stream>>>(x2, enr, ei, pE, eb1,
                                                 ew2, eb2, ew3, eb3, ew4, eb4, ew5, eb5,
                                                 out, E);
}
